// BEV_MSDA_58102317580777
// MI455X (gfx1250) — matrix operand order repaired and checked by reading, not compiled, not run on hardware
//
#include <hip/hip_runtime.h>
#include <hip/hip_bf16.h>
#include <math.h>

#define NQ     40000
#define DMODEL 256
#define NH     8
#define NPTS   6
#define HD     32
#define IMH    200
#define IMW    200
#define KDIM   256
#define RPW    8   // row tiles processed per wave in the GEMM

typedef __attribute__((ext_vector_type(16))) _Float16 v16h;
typedef __attribute__((ext_vector_type(8)))  float    v8f;

// Build one 16-bit A fragment (16x32 tile, this lane's half) from a row of A.
// Elems 0..7 -> K = base + h*8 + j ; elems 8..15 -> K = base + 16 + h*8 + j.
__device__ __forceinline__ v16h load_afrag(const float* __restrict__ ap)
{
    v16h a;
#pragma unroll
    for (int j = 0; j < 8; ++j) a[j]     = (_Float16)ap[j];
#pragma unroll
    for (int j = 0; j < 8; ++j) a[8 + j] = (_Float16)ap[16 + j];
    return a;
}

// -------------------------------------------------------------------------
// Generic C[N,M] = A[N,256] @ W[256,M] + bias  (fp32 in/out, f16 WMMA, f32 acc)
// One wave owns one 16-col tile; B fragments for all K=256 live in registers.
// Row tiles are processed in PAIRS -> two independent WMMA accumulation
// chains per K-step to hide load latency and the WMMA RAW hazard.
// ACT==1 applies exact GELU in the epilogue.
// -------------------------------------------------------------------------
template<int ACT>
__global__ __launch_bounds__(256)
void gemm_k256_wmma(const float* __restrict__ A, const float* __restrict__ W,
                    const float* __restrict__ bias, float* __restrict__ C,
                    int N, int M)
{
    const int lane = threadIdx.x & 31;
    const int r    = lane & 15;   // A row within tile / B+C column within tile
    const int h    = lane >> 4;   // half select per ISA 16-bit layouts
    const int waveId = blockIdx.x * (blockDim.x >> 5) + (threadIdx.x >> 5);

    const int colTiles  = M >> 4;
    const int rowTiles  = N >> 4;
    const int rowGroups = (rowTiles + RPW - 1) / RPW;
    if (waveId >= colTiles * rowGroups) return;

    const int colTile  = waveId % colTiles;
    const int rowGroup = waveId / colTiles;
    const int c0 = colTile * 16 + r;          // this lane's output column

    // B fragments: 32x16 tile per K-step; elems 0..7 hold K=kk+h*8+j,
    // elems 8..15 hold K=kk+16+h*8+j, column = lane&15.
    v16h bfrag[8];
#pragma unroll
    for (int kb = 0; kb < 8; ++kb) {
#pragma unroll
        for (int i = 0; i < 16; ++i) {
            const int k = kb * 32 + (i < 8 ? h * 8 + i : 16 + h * 8 + (i - 8));
            bfrag[kb][i] = (_Float16)W[(size_t)k * M + c0];
        }
    }

    const float bia   = bias[c0];
    const int   rtBeg = rowGroup * RPW;
    const int   rtEnd = min(rtBeg + RPW, rowTiles);

    int rt = rtBeg;
    // ---- paired row tiles: two independent accumulation chains ----
    for (; rt + 1 < rtEnd; rt += 2) {
        const float* arow0 = A + (size_t)(rt * 16 + r) * KDIM;
        const float* arow1 = arow0 + (size_t)16 * KDIM;

        // Warm the cache for the next pair (speculative; OOB is dropped).
        __builtin_prefetch(arow0 + (size_t)32 * KDIM, 0, 3);
        __builtin_prefetch(arow1 + (size_t)32 * KDIM, 0, 3);

        v8f acc0 = {}, acc1 = {};
#pragma unroll
        for (int kb = 0; kb < 8; ++kb) {
            const v16h a0 = load_afrag(arow0 + kb * 32 + h * 8);
            const v16h a1 = load_afrag(arow1 + kb * 32 + h * 8);
            acc0 = __builtin_amdgcn_wmma_f32_16x16x32_f16(
                       false, a0, false, bfrag[kb], (short)0, acc0, false, false);
            acc1 = __builtin_amdgcn_wmma_f32_16x16x32_f16(
                       false, a1, false, bfrag[kb], (short)0, acc1, false, false);
        }
#pragma unroll
        for (int i = 0; i < 8; ++i) {
            const int row0 = rt * 16 + h * 8 + i;        // C/D layout: row = h*8+i
            const int row1 = row0 + 16;
            float v0 = acc0[i] + bia;
            float v1 = acc1[i] + bia;
            if (ACT == 1) {
                v0 = 0.5f * v0 * (1.0f + erff(v0 * 0.70710678118654752f));
                v1 = 0.5f * v1 * (1.0f + erff(v1 * 0.70710678118654752f));
            }
            C[(size_t)row0 * M + c0] = v0;
            C[(size_t)row1 * M + c0] = v1;
        }
    }
    // ---- remainder single row tile ----
    if (rt < rtEnd) {
        const float* arow = A + (size_t)(rt * 16 + r) * KDIM;
        v8f acc = {};
#pragma unroll
        for (int kb = 0; kb < 8; ++kb) {
            const v16h a0 = load_afrag(arow + kb * 32 + h * 8);
            acc = __builtin_amdgcn_wmma_f32_16x16x32_f16(
                      false, a0, false, bfrag[kb], (short)0, acc, false, false);
        }
#pragma unroll
        for (int i = 0; i < 8; ++i) {
            const int row = rt * 16 + h * 8 + i;
            float vv = acc[i] + bia;
            if (ACT == 1) vv = 0.5f * vv * (1.0f + erff(vv * 0.70710678118654752f));
            C[(size_t)row * M + c0] = vv;
        }
    }
}

// -------------------------------------------------------------------------
// MSDA core: softmax over 6 points + bilinear gather + weighted sum.
// One wave per (query, head); lane = channel (HD=32 == wave32).
// v layout: [H*W, NH*HD]; out layout: [NQ, NH*HD].
// -------------------------------------------------------------------------
__global__ __launch_bounds__(256)
void msda_sample(const float* __restrict__ v, const float* __restrict__ off,
                 const float* __restrict__ attnlog, const float* __restrict__ refp,
                 float* __restrict__ out)
{
    const int waveId = blockIdx.x * (blockDim.x >> 5) + (threadIdx.x >> 5);
    if (waveId >= NQ * NH) return;
    const int q  = waveId / NH;
    const int hh = waveId % NH;
    const int c  = threadIdx.x & 31;

    // softmax over the 6 logits (same values in every lane -> broadcast loads)
    const float* al = attnlog + (size_t)q * (NH * NPTS) + hh * NPTS;
    float lg[NPTS], m = -1e30f;
#pragma unroll
    for (int p = 0; p < NPTS; ++p) { lg[p] = al[p]; m = fmaxf(m, lg[p]); }
    float s = 0.f;
#pragma unroll
    for (int p = 0; p < NPTS; ++p) { lg[p] = expf(lg[p] - m); s += lg[p]; }
    const float inv = 1.0f / s;

    const float rx = refp[(size_t)q * 2 + 0];
    const float ry = refp[(size_t)q * 2 + 1];
    const float* ofp = off + (size_t)q * (NH * NPTS * 2) + hh * (NPTS * 2);

    float acc = 0.f;
#pragma unroll
    for (int p = 0; p < NPTS; ++p) {
        // loc = ref + off / [W, H];  lx = loc.x*W - 0.5, ly = loc.y*H - 0.5
        const float lx = (rx + ofp[p * 2 + 0] * (1.0f / IMW)) * IMW - 0.5f;
        const float ly = (ry + ofp[p * 2 + 1] * (1.0f / IMH)) * IMH - 0.5f;
        const float fx = floorf(lx), fy = floorf(ly);
        const int x0 = (int)fx, y0 = (int)fy;
        const float tx = lx - fx, ty = ly - fy;
        const float wp = lg[p] * inv;
#pragma unroll
        for (int dy = 0; dy < 2; ++dy) {
#pragma unroll
            for (int dx = 0; dx < 2; ++dx) {
                const int xi = x0 + dx, yi = y0 + dy;
                const float w = (dx ? tx : 1.0f - tx) * (dy ? ty : 1.0f - ty);
                const bool valid = (xi >= 0) && (xi < IMW) && (yi >= 0) && (yi < IMH);
                const int xc = min(max(xi, 0), IMW - 1);
                const int yc = min(max(yi, 0), IMH - 1);
                const float g = v[(size_t)(yc * IMW + xc) * DMODEL + hh * HD + c];
                acc += valid ? wp * w * g : 0.0f;
            }
        }
    }
    out[(size_t)q * DMODEL + hh * HD + c] = acc;
}

// -------------------------------------------------------------------------
// out[row] = LayerNorm(x[row] + res[row]) * g + b   (one wave per row)
// -------------------------------------------------------------------------
__global__ __launch_bounds__(256)
void residual_ln(const float* __restrict__ x, const float* __restrict__ res,
                 const float* __restrict__ g, const float* __restrict__ b,
                 float* __restrict__ out, int N)
{
    const int waveId = blockIdx.x * (blockDim.x >> 5) + (threadIdx.x >> 5);
    if (waveId >= N) return;
    const int lane = threadIdx.x & 31;
    const float* xr = x   + (size_t)waveId * DMODEL;
    const float* rr = res + (size_t)waveId * DMODEL;

    float vals[8];
    float s = 0.f, s2 = 0.f;
#pragma unroll
    for (int j = 0; j < 8; ++j) {
        const float t = xr[lane + 32 * j] + rr[lane + 32 * j];
        vals[j] = t; s += t; s2 += t * t;
    }
#pragma unroll
    for (int msk = 16; msk >= 1; msk >>= 1) {
        s  += __shfl_xor(s,  msk, 32);
        s2 += __shfl_xor(s2, msk, 32);
    }
    const float mean = s * (1.0f / DMODEL);
    const float var  = s2 * (1.0f / DMODEL) - mean * mean;
    const float rstd = rsqrtf(var + 1e-5f);
#pragma unroll
    for (int j = 0; j < 8; ++j) {
        const int col = lane + 32 * j;
        out[(size_t)waveId * DMODEL + col] = (vals[j] - mean) * rstd * g[col] + b[col];
    }
}

// -------------------------------------------------------------------------
static inline void launch_gemm(int act, const float* A, const float* W,
                               const float* bias, float* C, int N, int M,
                               hipStream_t stream)
{
    const int colTiles  = M >> 4;
    const int rowTiles  = N >> 4;
    const int rowGroups = (rowTiles + RPW - 1) / RPW;
    const int totalWaves = colTiles * rowGroups;
    const int blocks = (totalWaves + 7) / 8;
    if (act) gemm_k256_wmma<1><<<blocks, 256, 0, stream>>>(A, W, bias, C, N, M);
    else     gemm_k256_wmma<0><<<blocks, 256, 0, stream>>>(A, W, bias, C, N, M);
}

extern "C" void kernel_launch(void* const* d_in, const int* in_sizes, int n_in,
                              void* d_out, int out_size, void* d_ws, size_t ws_size,
                              hipStream_t stream)
{
    (void)in_sizes; (void)n_in; (void)out_size; (void)ws_size;

    const float* query  = (const float*)d_in[0];
    const float* value  = (const float*)d_in[1];
    const float* refp   = (const float*)d_in[2];
    // d_in[3]=spatial_shapes, d_in[4]=level_start_index: single level 200x200, hardcoded
    const float* W_off  = (const float*)d_in[5];
    const float* b_off  = (const float*)d_in[6];
    const float* W_attn = (const float*)d_in[7];
    const float* b_attn = (const float*)d_in[8];
    const float* W_val  = (const float*)d_in[9];
    const float* b_val  = (const float*)d_in[10];
    const float* W_out  = (const float*)d_in[11];
    const float* b_out  = (const float*)d_in[12];
    const float* ln1_g  = (const float*)d_in[13];
    const float* ln1_b  = (const float*)d_in[14];
    const float* W1     = (const float*)d_in[15];
    const float* b1     = (const float*)d_in[16];
    const float* W2     = (const float*)d_in[17];
    const float* b2     = (const float*)d_in[18];
    const float* ln2_g  = (const float*)d_in[19];
    const float* ln2_b  = (const float*)d_in[20];

    float* ws = (float*)d_ws;
    const size_t NQD = (size_t)NQ * DMODEL;
    float* v_buf    = ws;                              // [40000,256]
    float* off_buf  = v_buf    + NQD;                  // [40000,96]
    float* attn_buf = off_buf  + (size_t)NQ * 96;      // [40000,48]
    float* msda_buf = attn_buf + (size_t)NQ * 48;      // [40000,256]
    float* y_buf    = msda_buf + NQD;                  // [40000,256]
    float* ymsda    = (float*)d_out;                   // staged; final LN overwrites
    float* h_buf    = v_buf;                           // reuse after sampling
    float* z_buf    = msda_buf;                        // reuse after out-proj

    // Projections (query -> sampling offsets / attention logits; value -> v)
    launch_gemm(0, query, W_off,  b_off,  off_buf,  NQ,        96,  stream);
    launch_gemm(0, query, W_attn, b_attn, attn_buf, NQ,        48,  stream);
    launch_gemm(0, value, W_val,  b_val,  v_buf,    IMH * IMW, 256, stream);

    // Deformable sampling core
    {
        const int totalWaves = NQ * NH;
        const int blocks = (totalWaves + 7) / 8;
        msda_sample<<<blocks, 256, 0, stream>>>(v_buf, off_buf, attn_buf, refp, msda_buf);
    }

    // Output projection + residual LayerNorm 1
    launch_gemm(0, msda_buf, W_out, b_out, ymsda, NQ, 256, stream);
    {
        const int blocks = (NQ + 7) / 8;
        residual_ln<<<blocks, 256, 0, stream>>>(ymsda, query, ln1_g, ln1_b, y_buf, NQ);
    }

    // FFN: GELU(y@W1+b1)@W2+b2, then residual LayerNorm 2 -> d_out
    launch_gemm(1, y_buf, W1, b1, h_buf, NQ, 256, stream);
    launch_gemm(0, h_buf, W2, b2, z_buf, NQ, 256, stream);
    {
        const int blocks = (NQ + 7) / 8;
        residual_ln<<<blocks, 256, 0, stream>>>(z_buf, y_buf, ln2_g, ln2_b,
                                                (float*)d_out, NQ);
    }
}
